// discriminator_42872363548906
// MI455X (gfx1250) — compile-verified, matrix operand order checked by reading, not run on hardware
//
#include <hip/hip_runtime.h>
#include <math.h>

typedef _Float16 half_t;
typedef __attribute__((ext_vector_type(16))) _Float16 v16h;
typedef __attribute__((ext_vector_type(8)))  float    v8f;

// ---------------------------------------------------------------------------
// Elementwise / data-movement kernels
// ---------------------------------------------------------------------------

// points (B,14,N) -> pts (B,N,14), xyz (B,N,3)
static __global__ void k_transpose(const float* __restrict__ in, float* __restrict__ pts,
                                   float* __restrict__ xyz, int total, int N) {
  int i = blockIdx.x * blockDim.x + threadIdx.x;
  if (i >= total) return;
  int b = i / N, n = i % N;
  const float* src = in + (size_t)b * 14 * N + n;
  float* dp = pts + (size_t)i * 14;
#pragma unroll
  for (int c = 0; c < 14; ++c) dp[c] = src[(size_t)c * N];
  float* dx = xyz + (size_t)i * 3;
  dx[0] = dp[0]; dx[1] = dp[1]; dx[2] = dp[2];
}

// Farthest point sampling: one block (256 threads) per batch. Sequential over
// npoint iterations, LDS argmax tree each iteration (first-max tie-break).
static __global__ void k_fps(const float* __restrict__ xyz, int n, int npoint,
                             int* __restrict__ out) {
  int b = blockIdx.x, t = threadIdx.x;
  __shared__ float dist[8192];
  __shared__ float rv[256];
  __shared__ int   ri[256];
  __shared__ int   cur;
  const float* base = xyz + (size_t)b * n * 3;
  for (int i = t; i < n; i += blockDim.x) dist[i] = 1e10f;
  if (t == 0) cur = 0;
  __syncthreads();
  for (int it = 0; it < npoint; ++it) {
    int far = cur;
    if (t == 0) out[b * npoint + it] = far;
    float cx = base[far * 3 + 0], cy = base[far * 3 + 1], cz = base[far * 3 + 2];
    float bv = -1.0f; int bi = n - 1;
    for (int i = t; i < n; i += blockDim.x) {
      float dx = base[i * 3 + 0] - cx;
      float dy = base[i * 3 + 1] - cy;
      float dz = base[i * 3 + 2] - cz;
      float d = dx * dx + dy * dy + dz * dz;
      float nd = fminf(dist[i], d);
      dist[i] = nd;
      if (nd > bv || (nd == bv && i < bi)) { bv = nd; bi = i; }
    }
    rv[t] = bv; ri[t] = bi;
    __syncthreads();
    for (int s = 128; s > 0; s >>= 1) {
      if (t < s) {
        if (rv[t + s] > rv[t] || (rv[t + s] == rv[t] && ri[t + s] < ri[t])) {
          rv[t] = rv[t + s]; ri[t] = ri[t + s];
        }
      }
      __syncthreads();
    }
    if (t == 0) cur = ri[0];
    __syncthreads();
  }
}

// gather xyz rows by index: out[b,i,:] = xyz[b, idx[b,i], :]
static __global__ void k_gather3(const float* __restrict__ xyz, const int* __restrict__ idx,
                                 float* __restrict__ out, int n, int total) {
  int i = blockIdx.x * blockDim.x + threadIdx.x;
  if (i >= total) return;
  int S = total / 4;                 // total = B*S
  int b = i / S;
  int j = idx[i];
  const float* p = xyz + ((size_t)b * n + j) * 3;
  float* o = out + (size_t)i * 3;
  o[0] = p[0]; o[1] = p[1]; o[2] = p[2];
}

// ball query: first k ascending in-radius indices, pad with first (clamped)
static __global__ void k_query_ball(const float* __restrict__ xyz, const float* __restrict__ nxyz,
                                    int n, int S, float r2, int k, int* __restrict__ gidx,
                                    int total) {
  int i = blockIdx.x * blockDim.x + threadIdx.x;
  if (i >= total) return;     // total = B*S
  int b = i / S;
  const float* c = nxyz + (size_t)i * 3;
  float cx = c[0], cy = c[1], cz = c[2];
  const float* base = xyz + (size_t)b * n * 3;
  int* out = gidx + (size_t)i * k;
  int cnt = 0, first = -1;
  for (int j = 0; j < n && cnt < k; ++j) {
    float dx = base[j * 3 + 0] - cx;
    float dy = base[j * 3 + 1] - cy;
    float dz = base[j * 3 + 2] - cz;
    float d = dx * dx + dy * dy + dz * dz;
    if (d <= r2) { if (first < 0) first = j; out[cnt++] = j; }
  }
  if (first < 0) first = n - 1;          // JAX OOB-gather clamp semantics
  for (; cnt < k; ++cnt) out[cnt] = first;
}

// build grouped activation rows in f16: G[row,0:cin]=feats[j],
// G[row,cin:cin+3]=xyz[j]-new_xyz, zero-pad to Kp (K-padded for WMMA GEMM).
static __global__ void k_group(const float* __restrict__ feats, const float* __restrict__ xyz,
                               const float* __restrict__ nxyz, const int* __restrict__ gidx,
                               int n, int S, int k, int cin, int Kp,
                               half_t* __restrict__ G, int total) {
  int r = blockIdx.x * blockDim.x + threadIdx.x;
  if (r >= total) return;                 // total = B*S*k
  int b = r / (S * k);
  int rem = r % (S * k);
  int si = rem / k;
  int j = gidx[r];
  const float* f = feats + ((size_t)b * n + j) * cin;
  half_t* g = G + (size_t)r * Kp;
  for (int c = 0; c < cin; ++c) g[c] = (half_t)f[c];
  const float* p = xyz + ((size_t)b * n + j) * 3;
  const float* q = nxyz + ((size_t)b * S + si) * 3;
  g[cin + 0] = (half_t)(p[0] - q[0]);
  g[cin + 1] = (half_t)(p[1] - q[1]);
  g[cin + 2] = (half_t)(p[2] - q[2]);
  for (int c = cin + 3; c < Kp; ++c) g[c] = (half_t)0.0f;
}

// W (K,N) fp32 -> Wt (N,Kp) f16, zero-padded (WMMA-B-friendly transposed layout)
static __global__ void k_pack_wt(const float* __restrict__ W, int K, int Nc, int Kp,
                                 half_t* __restrict__ Wt) {
  int i = blockIdx.x * blockDim.x + threadIdx.x;
  if (i >= Nc * Kp) return;
  int n = i / Kp, kk = i % Kp;
  Wt[i] = (kk < K) ? (half_t)W[(size_t)kk * Nc + n] : (half_t)0.0f;
}

// ---------------------------------------------------------------------------
// WMMA GEMM: Y[M,Nc] = A[M,Kp] @ Wt^T + bias.  A f16 (Kp-padded, zero fill),
// Wt f16 (N x Kp).  Each wave owns a 64x16 output strip: 4 M-subtiles share
// one B fragment per K-chunk -> 4x v_wmma_f32_16x16x32_f16 per chunk, pure
// b128 f16 loads (no inner-loop cvt).
// Fragment layout per CDNA5 ISA 7.12.2:
//   lanes 0-15 : halves 0..7 -> K 0..7,  halves 8..15 -> K 16..23
//   lanes 16-31: halves 0..7 -> K 8..15, halves 8..15 -> K 24..31
//   D: vgpr r, lane<16 -> (M=r, N=lane); lane>=16 -> (M=r+8, N=lane-16)
// ---------------------------------------------------------------------------
static __global__ void k_gemm(const half_t* __restrict__ A, const half_t* __restrict__ Wt,
                              const float* __restrict__ bias, float* __restrict__ Y,
                              int M, int Nc, int Kp) {
  int lane  = threadIdx.x;                          // 0..31 (wave32)
  int tileN = blockIdx.y;
  int tm0   = (blockIdx.x * blockDim.y + threadIdx.y) * 4;  // first of 4 M-tiles
  int col = tileN * 16 + (lane & 15);
  int cc  = (col < Nc) ? col : (Nc - 1);
  const half_t* pb = Wt + (size_t)cc * Kp;
  int q8 = (lane >> 4) * 8;

  const half_t* pa[4];
#pragma unroll
  for (int t = 0; t < 4; ++t) {
    int row = (tm0 + t) * 16 + (lane & 15);
    int rc  = (row < M) ? row : (M - 1);            // clamp; OOB results discarded
    pa[t] = A + (size_t)rc * Kp;
  }
  v8f acc0 = {}, acc1 = {}, acc2 = {}, acc3 = {};
  for (int kb = 0; kb < Kp; kb += 32) {
    __builtin_prefetch(pa[0] + kb + 128, 0, 1);     // global_prefetch_b8 (CDNA5)
    v16h bv;
#pragma unroll
    for (int h = 0; h < 8; ++h) {
      bv[h]     = pb[kb + q8 + h];
      bv[8 + h] = pb[kb + 16 + q8 + h];
    }
    v16h a0, a1, a2, a3;
#pragma unroll
    for (int h = 0; h < 8; ++h) {
      a0[h] = pa[0][kb + q8 + h];  a0[8 + h] = pa[0][kb + 16 + q8 + h];
      a1[h] = pa[1][kb + q8 + h];  a1[8 + h] = pa[1][kb + 16 + q8 + h];
      a2[h] = pa[2][kb + q8 + h];  a2[8 + h] = pa[2][kb + 16 + q8 + h];
      a3[h] = pa[3][kb + q8 + h];  a3[8 + h] = pa[3][kb + 16 + q8 + h];
    }
    acc0 = __builtin_amdgcn_wmma_f32_16x16x32_f16(false, a0, false, bv, (short)0, acc0, false, false);
    acc1 = __builtin_amdgcn_wmma_f32_16x16x32_f16(false, a1, false, bv, (short)0, acc1, false, false);
    acc2 = __builtin_amdgcn_wmma_f32_16x16x32_f16(false, a2, false, bv, (short)0, acc2, false, false);
    acc3 = __builtin_amdgcn_wmma_f32_16x16x32_f16(false, a3, false, bv, (short)0, acc3, false, false);
  }
  if (col < Nc) {
    float bb = bias ? bias[col] : 0.0f;
#pragma unroll
    for (int r = 0; r < 8; ++r) {
      int orow;
      orow = (tm0 + 0) * 16 + r + q8; if (orow < M) Y[(size_t)orow * Nc + col] = acc0[r] + bb;
      orow = (tm0 + 1) * 16 + r + q8; if (orow < M) Y[(size_t)orow * Nc + col] = acc1[r] + bb;
      orow = (tm0 + 2) * 16 + r + q8; if (orow < M) Y[(size_t)orow * Nc + col] = acc2[r] + bb;
      orow = (tm0 + 3) * 16 + r + q8; if (orow < M) Y[(size_t)orow * Nc + col] = acc3[r] + bb;
    }
  }
}

// per-channel sum & sumsq over M rows: one block per channel, deterministic
static __global__ void k_bn_stats(const float* __restrict__ Y, int M, int Nc,
                                  float* __restrict__ stats) {
  int c = blockIdx.x, t = threadIdx.x;
  float s = 0.0f, s2 = 0.0f;
  for (int r = t; r < M; r += blockDim.x) {
    float v = Y[(size_t)r * Nc + c];
    s += v; s2 += v * v;
  }
  __shared__ float sh[256], sh2[256];
  sh[t] = s; sh2[t] = s2;
  __syncthreads();
  for (int w = 128; w > 0; w >>= 1) {
    if (t < w) { sh[t] += sh[t + w]; sh2[t] += sh2[t + w]; }
    __syncthreads();
  }
  if (t == 0) { stats[c] = sh[0]; stats[Nc + c] = sh2[0]; }
}

// BN (training-mode, population var) + ReLU -> f16, ldo-strided with zero
// K-padding so the result feeds the next WMMA GEMM directly.
static __global__ void k_bn_relu_h(const float* __restrict__ Y, const float* __restrict__ stats,
                                   const float* __restrict__ gamma, const float* __restrict__ beta,
                                   int M, int Nc, int ldo, float invM, half_t* __restrict__ out,
                                   long long total) {
  long long i = (long long)blockIdx.x * blockDim.x + threadIdx.x;
  if (i >= total) return;
  int c = (int)(i % ldo);
  long long r = i / ldo;
  if (c < Nc) {
    float mu  = stats[c] * invM;
    float var = stats[Nc + c] * invM - mu * mu;
    float v = (Y[(size_t)r * Nc + c] - mu) * rsqrtf(var + 1e-5f) * gamma[c] + beta[c];
    out[i] = (half_t)(v > 0.0f ? v : 0.0f);
  } else {
    out[i] = (half_t)0.0f;
  }
}

// BN + ReLU -> fp32 unpadded (for max-pool / persisted feature tensors)
static __global__ void k_bn_relu_f(const float* __restrict__ Y, const float* __restrict__ stats,
                                   const float* __restrict__ gamma, const float* __restrict__ beta,
                                   int M, int Nc, float invM, float* __restrict__ out,
                                   long long total) {
  long long i = (long long)blockIdx.x * blockDim.x + threadIdx.x;
  if (i >= total) return;
  int c = (int)(i % Nc);
  long long r = i / Nc;
  float mu  = stats[c] * invM;
  float var = stats[Nc + c] * invM - mu * mu;
  float v = (Y[(size_t)r * Nc + c] - mu) * rsqrtf(var + 1e-5f) * gamma[c] + beta[c];
  out[i] = v > 0.0f ? v : 0.0f;
}

// max over the k neighbor axis; scatter into channel-offset slice of feats_out
static __global__ void k_maxpool(const float* __restrict__ A, int k, int cl,
                                 float* __restrict__ out, int ldout, int coff,
                                 long long total) {
  long long i = (long long)blockIdx.x * blockDim.x + threadIdx.x;
  if (i >= total) return;                   // total = B*S*cl
  long long row = i / cl;
  int cc = (int)(i % cl);
  const float* a = A + ((size_t)row * k) * cl + cc;
  float m = -3.4e38f;
  for (int j = 0; j < k; ++j) {
    float v = a[(size_t)j * cl];
    m = v > m ? v : m;
  }
  out[(size_t)row * ldout + coff + cc] = m;
}

// 3-NN inverse-distance interpolation -> f16 slice X[:, coff:coff+c2]
static __global__ void k_interp(const float* __restrict__ xyz1, const float* __restrict__ xyz2,
                                const float* __restrict__ p2, int total, int n1, int n2,
                                int c2, half_t* __restrict__ X, int ldx, int coff) {
  int i = blockIdx.x * blockDim.x + threadIdx.x;
  if (i >= total) return;                   // total = B*n1
  int b = i / n1;
  const float* p = xyz1 + (size_t)i * 3;
  float px = p[0], py = p[1], pz = p[2];
  const float* q = xyz2 + (size_t)b * n2 * 3;
  float d0 = 3.4e38f, d1 = 3.4e38f, d2v = 3.4e38f;
  int i0 = 0, i1 = 0, i2 = 0;
  for (int j = 0; j < n2; ++j) {
    float dx = q[j * 3 + 0] - px;
    float dy = q[j * 3 + 1] - py;
    float dz = q[j * 3 + 2] - pz;
    float d = dx * dx + dy * dy + dz * dz;
    if (d < d0)      { d2v = d1; i2 = i1; d1 = d0; i1 = i0; d0 = d; i0 = j; }
    else if (d < d1) { d2v = d1; i2 = i1; d1 = d;  i1 = j; }
    else if (d < d2v){ d2v = d;  i2 = j; }
  }
  float w0 = 1.0f / (d0 + 1e-8f), w1 = 1.0f / (d1 + 1e-8f), w2 = 1.0f / (d2v + 1e-8f);
  float ws = w0 + w1 + w2;
  w0 /= ws; w1 /= ws; w2 /= ws;
  const float* f0 = p2 + ((size_t)b * n2 + i0) * c2;
  const float* f1 = p2 + ((size_t)b * n2 + i1) * c2;
  const float* f2 = p2 + ((size_t)b * n2 + i2) * c2;
  half_t* dst = X + (size_t)i * ldx + coff;
  for (int c = 0; c < c2; ++c)
    dst[c] = (half_t)(w0 * f0[c] + w1 * f1[c] + w2 * f2[c]);
}

// copy p1 channels [0,c1) (fp32 -> f16) and zero pad columns [ctotal, ldx)
static __global__ void k_copy_pad(const float* __restrict__ p1, int total, int c1,
                                  half_t* __restrict__ X, int ldx, int ctotal) {
  int i = blockIdx.x * blockDim.x + threadIdx.x;
  if (i >= total) return;
  half_t* dst = X + (size_t)i * ldx;
  if (p1) {
    const float* s = p1 + (size_t)i * c1;
    for (int c = 0; c < c1; ++c) dst[c] = (half_t)s[c];
  }
  for (int c = ctotal; c < ldx; ++c) dst[c] = (half_t)0.0f;
}

// head: out[i] = sigmoid(A[i,:32] . W2 + b2)
static __global__ void k_head_out(const float* __restrict__ A, const float* __restrict__ W2,
                                  const float* __restrict__ b2, float* __restrict__ out, int M) {
  int i = blockIdx.x * blockDim.x + threadIdx.x;
  if (i >= M) return;
  const float* a = A + (size_t)i * 32;
  float s = b2[0];
#pragma unroll
  for (int c = 0; c < 32; ++c) s += a[c] * W2[c];
  out[i] = 1.0f / (1.0f + expf(-s));
}

// ---------------------------------------------------------------------------
// Host orchestration
// ---------------------------------------------------------------------------
struct Lyr { const float* W; const float* b; const float* g; const float* be; int cin, cout; };

extern "C" void kernel_launch(void* const* d_in, const int* in_sizes, int n_in,
                              void* d_out, int out_size, void* d_ws, size_t ws_size,
                              hipStream_t stream) {
  (void)in_sizes; (void)n_in; (void)out_size; (void)ws_size;
  const int B = 4, N = 8192;
  static const int   sa_nscale[4]  = {3, 2, 2, 2};
  static const float sa_radii[4][3]= {{0.02f,0.05f,0.1f},{0.1f,0.2f,0.f},{0.2f,0.4f,0.f},{0.4f,0.8f,0.f}};
  static const int   sa_ks[4][3]   = {{8,16,32},{16,32,0},{16,32,0},{16,32,0}};
  static const int   sa_cin[4]     = {14, 96, 128, 192};
  static const int   sa_mlps[4][3][3] = {
      {{8,8,16},{16,16,32},{32,32,48}},
      {{48,48,64},{48,48,64},{0,0,0}},
      {{64,64,96},{64,64,96},{0,0,0}},
      {{96,96,128},{96,96,128},{0,0,0}}};
  const int npts[5] = {8192, 2048, 512, 128, 32};
  const int nch[5]  = {14, 96, 128, 192, 256};

  // ---- walk the flattened parameter pytree (setup_inputs dict order) ----
  int pi = 0;
  const float* d_points = (const float*)d_in[pi++];
  Lyr sa[4][3][3];
  for (int L = 0; L < 4; ++L)
    for (int s = 0; s < sa_nscale[L]; ++s) {
      int cin = sa_cin[L] + 3;
      for (int l = 0; l < 3; ++l) {
        Lyr& y = sa[L][s][l];
        y.W  = (const float*)d_in[pi++];
        y.b  = (const float*)d_in[pi++];
        y.g  = (const float*)d_in[pi++];
        y.be = (const float*)d_in[pi++];
        y.cin = cin; y.cout = sa_mlps[L][s][l];
        cin = y.cout;
      }
    }
  static const int fp_nl[4]   = {2, 2, 2, 3};
  static const int fp_cin[4]  = {448, 256, 224, 64};
  static const int fp_co[4][3]= {{128,128,0},{128,128,0},{128,64,0},{64,64,64}};
  Lyr fp[4][3];
  for (int m = 0; m < 4; ++m) {
    int cin = fp_cin[m];
    for (int l = 0; l < fp_nl[m]; ++l) {
      Lyr& y = fp[m][l];
      y.W  = (const float*)d_in[pi++];
      y.b  = (const float*)d_in[pi++];
      y.g  = (const float*)d_in[pi++];
      y.be = (const float*)d_in[pi++];
      y.cin = cin; y.cout = fp_co[m][l];
      cin = y.cout;
    }
  }
  const float* h_c1W = (const float*)d_in[pi++];
  const float* h_c1b = (const float*)d_in[pi++];
  const float* h_bg  = (const float*)d_in[pi++];
  const float* h_bb  = (const float*)d_in[pi++];
  const float* h_c2W = (const float*)d_in[pi++];
  const float* h_c2b = (const float*)d_in[pi++];

  // ---- workspace arena ----
  size_t off = 0;
  auto alloc = [&](size_t nbytes) -> void* {
    off = (off + 255) & ~(size_t)255;
    void* p = (char*)d_ws + off;
    off += nbytes;
    return p;
  };
  auto r32 = [](int x) { return (x + 31) & ~31; };

  // bufA: f16 activations (M x Kp) OR fp32 finals (M x Nc); bufB: fp32 GEMM out
  const size_t MAXBUF = (size_t)262144 * 64;  // elements; covers every tensor
  char*  bufA = (char*)alloc(MAXBUF * 4);
  float* bufB = (float*)alloc(MAXBUF * 4);
  float* pts  = (float*)alloc((size_t)B * N * 14 * 4);
  float* xyz[5];
  float* feats[5];
  xyz[0] = (float*)alloc((size_t)B * N * 3 * 4);
  feats[0] = pts;
  for (int L = 1; L < 5; ++L) {
    xyz[L]   = (float*)alloc((size_t)B * npts[L] * 3 * 4);
    feats[L] = (float*)alloc((size_t)B * npts[L] * nch[L] * 4);
  }
  float* l3buf = (float*)alloc((size_t)B * 128 * 128 * 4);
  float* l2buf = (float*)alloc((size_t)B * 512 * 128 * 4);
  float* l1buf = (float*)alloc((size_t)B * 2048 * 64 * 4);
  int*    fpsIdx = (int*)alloc((size_t)B * 2048 * 4);
  int*    gidx   = (int*)alloc((size_t)B * 2048 * 32 * 4);
  half_t* wtbuf  = (half_t*)alloc((size_t)128 * 448 * 2);
  float*  stats  = (float*)alloc(256 * 4);

  half_t* hA = (half_t*)bufA;
  float*  fA = (float*)bufA;

  const int T = 256;
  auto blocks = [&](long long n) { return (unsigned)((n + T - 1) / T); };

  // GEMM launcher: one wave (32 lanes) per 64x16 strip, 4 waves per block
  auto launch_gemm = [&](const half_t* A, const half_t* Wt, const float* bias,
                         float* Y, int M, int Nc, int Kp) {
    int tilesM = (M + 15) / 16;
    int strips = (tilesM + 3) / 4;            // 4 M-tiles per wave
    dim3 gg((strips + 3) / 4, (Nc + 15) / 16), bb(32, 4);
    k_gemm<<<gg, bb, 0, stream>>>(A, Wt, bias, Y, M, Nc, Kp);
  };

  // run an MLP chain in bufA (M x Kp0, K-padded f16). Inner layers write f16
  // padded back to bufA; final layer -> f16 padded (finalHalf) or fp32 ld=Nc.
  auto run_mlp = [&](const Lyr* Ls, int nl, int M, int Kp0, bool finalHalf) {
    int curKp = Kp0;
    for (int l = 0; l < nl; ++l) {
      int K = Ls[l].cin, Nc = Ls[l].cout;
      k_pack_wt<<<blocks((long long)Nc * curKp), T, 0, stream>>>(Ls[l].W, K, Nc, curKp, wtbuf);
      launch_gemm(hA, wtbuf, Ls[l].b, bufB, M, Nc, curKp);
      k_bn_stats<<<Nc, T, 0, stream>>>(bufB, M, Nc, stats);
      bool half_out = (l < nl - 1) || finalHalf;
      if (half_out) {
        int ldo = r32(Nc);
        long long tot = (long long)M * ldo;
        k_bn_relu_h<<<blocks(tot), T, 0, stream>>>(bufB, stats, Ls[l].g, Ls[l].be,
                                                   M, Nc, ldo, 1.0f / (float)M, hA, tot);
        curKp = ldo;
      } else {
        long long tot = (long long)M * Nc;
        k_bn_relu_f<<<blocks(tot), T, 0, stream>>>(bufB, stats, Ls[l].g, Ls[l].be,
                                                   M, Nc, 1.0f / (float)M, fA, tot);
        curKp = Nc;
      }
    }
  };

  // ---- stage 0: transpose ----
  k_transpose<<<blocks((long long)B * N), T, 0, stream>>>(d_points, pts, xyz[0], B * N, N);

  // ---- set-abstraction levels ----
  for (int L = 0; L < 4; ++L) {
    int n = npts[L], S = npts[L + 1];
    k_fps<<<B, T, 0, stream>>>(xyz[L], n, S, fpsIdx);
    k_gather3<<<blocks((long long)B * S), T, 0, stream>>>(xyz[L], fpsIdx, xyz[L + 1], n, B * S);
    int coff = 0;
    for (int s = 0; s < sa_nscale[L]; ++s) {
      int k = sa_ks[L][s];
      float r = sa_radii[L][s];
      k_query_ball<<<blocks((long long)B * S), T, 0, stream>>>(xyz[L], xyz[L + 1], n, S,
                                                               r * r, k, gidx, B * S);
      int M = B * S * k;
      int cin = sa_cin[L];
      int Kp = r32(cin + 3);
      k_group<<<blocks((long long)M), T, 0, stream>>>(feats[L], xyz[L], xyz[L + 1], gidx,
                                                      n, S, k, cin, Kp, hA, M);
      run_mlp(sa[L][s], 3, M, Kp, /*finalHalf=*/false);
      int cl = sa[L][s][2].cout;
      long long tot = (long long)B * S * cl;
      k_maxpool<<<blocks(tot), T, 0, stream>>>(fA, k, cl, feats[L + 1], nch[L + 1], coff, tot);
      coff += cl;
    }
  }

  // ---- feature propagation ----
  const int    fi1[4]   = {3, 2, 1, 0}, fi2[4] = {4, 3, 2, 1};
  const float* fp_p1[4] = {feats[3], feats[2], feats[1], nullptr};
  const int    fp_c1[4] = {192, 128, 96, 0};
  const float* fp_p2[4] = {feats[4], l3buf, l2buf, l1buf};
  const int    fp_c2[4] = {256, 128, 128, 64};
  float* fp_save[4] = {l3buf, l2buf, l1buf, nullptr};

  for (int m = 0; m < 4; ++m) {
    int n1 = npts[fi1[m]], n2 = npts[fi2[m]];
    int M = B * n1;
    int ctot = fp_cin[m];
    int Kp = r32(ctot);
    k_copy_pad<<<blocks((long long)M), T, 0, stream>>>(fp_p1[m], M, fp_c1[m], hA, Kp, ctot);
    k_interp<<<blocks((long long)M), T, 0, stream>>>(xyz[fi1[m]], xyz[fi2[m]], fp_p2[m],
                                                     M, n1, n2, fp_c2[m], hA, Kp, fp_c1[m]);
    // m<3: final fp32 (persisted, consumed by next interp); m==3: f16 for head
    run_mlp(fp[m], fp_nl[m], M, Kp, /*finalHalf=*/(m == 3));
    int cl = fp[m][fp_nl[m] - 1].cout;
    if (fp_save[m])
      hipMemcpyAsync(fp_save[m], fA, (size_t)M * cl * sizeof(float),
                     hipMemcpyDeviceToDevice, stream);
  }

  // ---- head: 64 -> 32 (BN+ReLU) -> 1 (sigmoid) ----
  {
    int M = B * N;                       // l0 in bufA as f16, ld = 64
    k_pack_wt<<<blocks((long long)32 * 64), T, 0, stream>>>(h_c1W, 64, 32, 64, wtbuf);
    launch_gemm(hA, wtbuf, h_c1b, bufB, M, 32, 64);
    k_bn_stats<<<32, T, 0, stream>>>(bufB, M, 32, stats);
    long long tot = (long long)M * 32;
    k_bn_relu_f<<<blocks(tot), T, 0, stream>>>(bufB, stats, h_bg, h_bb, M, 32,
                                               1.0f / (float)M, fA, tot);
    k_head_out<<<blocks((long long)M), T, 0, stream>>>(fA, h_c2W, h_c2b, (float*)d_out, M);
  }
}
